// ConvColumn_33337536152375
// MI455X (gfx1250) — matrix operand order repaired and checked by reading, not compiled, not run on hardware
//
#include <hip/hip_runtime.h>
#include <hip/hip_bf16.h>

typedef __attribute__((ext_vector_type(16))) _Float16 v16h;
typedef __attribute__((ext_vector_type(8)))  float    v8f;

// ---------------- problem constants ----------------
#define B_      16
#define CIN     2
#define COUT    32
#define HW      64
#define TIN     100
#define KSZ     3
#define KS      48          // temporal kernel length
#define KT_PAD  64          // padded temporal length per tap (multiple of 32)
#define NTAP    (CIN*KSZ*KSZ)            // 18
#define KTOT    (NTAP*KT_PAD)            // 1152
#define NCHUNK  (KTOT/32)                // 36
#define HP      31          // H' = (64-3)/2+1
#define NSPAT   (HP*HP)     // 961
#define TP      149         // T' = 100 + 2*48 - 48 + 1
#define NT      5           // t-tiles per wave
#define TGROUPS 2           // 2 groups of 5 tiles cover 10 tiles (t' 0..159)
#define TPAD    224         // padded time row length (halfs), even
#define THETA   5.4f
#define FODEP   48

// halfs per copy of padded input
#define CSTRIDE ((size_t)B_*CIN*HW*HW*TPAD)          // 29,360,128 halfs
// ws layout (bytes)
#define AW_OFF   0ull
#define XH_OFF   131072ull
#define XH_BYTES (2ull*CSTRIDE*2ull)                 // 117,440,512
#define POT_OFF  (XH_OFF + XH_BYTES)                 // 117,571,584
#define POT_BYTES ((size_t)B_*NSPAT*TP*COUT*4ull)    // 293,251,072
#define WS_NEED  (POT_OFF + POT_BYTES)

// ---------------- kernel 1: build A fragments (weights -> fp16, fragment layout) ----------------
// aw[mtile][chunk][lane][j], j = half index inside v16h
__global__ __launch_bounds__(256) void sfl_build_wk(const float* __restrict__ weight,
                                                    _Float16* __restrict__ aw) {
    int idx = blockIdx.x * 256 + threadIdx.x;            // 2*36*32*16 = 36864 total
    if (idx >= 2 * NCHUNK * 32 * 16) return;
    int j     = idx & 15;
    int lane  = (idx >> 4) & 31;
    int chunk = (idx >> 9) % NCHUNK;
    int mtile = idx / (NCHUNK * 512);
    int o = mtile * 16 + (lane & 15);
    // A-fragment k order used here (contiguous, same as B): K = j + 16*(lane/16)
    int Klocal = j + 16 * (lane >> 4);
    int kt  = ((chunk & 1) << 5) + Klocal;               // 0..63 within padded tap
    int tap = chunk >> 1;
    float val = 0.0f;
    if (kt < KS) {
        int ci = tap / 9, kh = (tap % 9) / 3, kw = tap % 3;
        float w = weight[((o * CIN + ci) * KSZ + kh) * KSZ + kw];
        float t = (float)(KS - 1 - kt);                  // time-flipped kernel
        float leak  = 1.5f * w - t * (1.0f / 32.0f);
        float spike = t * (1.0f / 16.0f);
        val = fminf(spike, fmaxf(0.0f, leak));
    }
    aw[idx] = (_Float16)val;
}

// ---------------- kernel 2: pad + convert input to fp16, two parity-shifted copies ----------------
// xh[p][b][i][h][w][tp] = x[t_in = tp + p - 48]  (0 outside [0,100))
__global__ __launch_bounds__(256) void sfl_pad_input(const float* __restrict__ x,
                                                     _Float16* __restrict__ xh,
                                                     long long total) {
    long long idx = (long long)blockIdx.x * 256 + threadIdx.x;
    if (idx >= total) return;
    long long t = idx;
    int tp = (int)(t % TPAD); t /= TPAD;
    int w  = (int)(t % HW);   t /= HW;
    int h  = (int)(t % HW);   t /= HW;
    int ci = (int)(t % CIN);  t /= CIN;
    int b  = (int)(t % B_);   t /= B_;
    int p  = (int)t;                                     // copy 0 / 1
    int tin = tp + p - KS;
    float v = 0.0f;
    if (tin >= 0 && tin < TIN)
        v = x[((((size_t)b * CIN + ci) * HW + h) * HW + w) * TIN + tin];
    xh[idx] = (_Float16)v;
}

// ---------------- kernel 3: zero output ----------------
__global__ __launch_bounds__(256) void sfl_zero(float4* __restrict__ out, int n4) {
    int i = blockIdx.x * 256 + threadIdx.x;
    if (i < n4) out[i] = make_float4(0.f, 0.f, 0.f, 0.f);
}

// ---------------- kernel 4: implicit-GEMM conv with WMMA ----------------
// one wave per (b, spatial, t-group); M=32 channels (2 tiles), K=1152,
// N = 5 tiles of 16 consecutive t' values (A fragments reused x5)
__global__ __launch_bounds__(256) void sfl_conv_wmma(const _Float16* __restrict__ xh,
                                                     const _Float16* __restrict__ aw,
                                                     float* __restrict__ pot) {
    int lane = threadIdx.x & 31;
    int wave = threadIdx.x >> 5;
    int tile = blockIdx.x * 8 + wave;                    // exact grid: 30,752 waves
    int g    = tile & 1;                                 // t-group: t' base 0 or 80
    int rest = tile >> 1;
    int s    = rest % NSPAT;
    int b    = rest / NSPAT;
    int hp = s / HP, wp = s % HP;
    int t0 = g * (NT * 16);
    int par = lane & 1;
    // copy `par`, read from (S - par): 4-byte aligned contiguous 32B per lane
    const _Float16* xbase = xh + (size_t)par * CSTRIDE - (size_t)par;

    v8f acc[2 * NT];
#pragma unroll
    for (int q = 0; q < 2 * NT; ++q) acc[q] = (v8f){0.f,0.f,0.f,0.f,0.f,0.f,0.f,0.f};

    // base row offset (halfs) for ci=0, kh=0, kw=0
    size_t base0 = (((size_t)(b * CIN) * HW + 2 * hp) * HW + 2 * wp) * TPAD
                   + (size_t)(t0 + lane);
    const _Float16* awp = aw;                            // advances 512 halfs per chunk

    int ci = 0, kh = 0, kw = 0;
    for (int tap = 0; tap < NTAP; ++tap) {
        size_t rowS = base0 + (size_t)ci * ((size_t)HW * HW * TPAD)
                            + (size_t)kh * ((size_t)HW * TPAD)
                            + (size_t)kw * (size_t)TPAD;
#pragma unroll
        for (int half = 0; half < 2; ++half) {
            const v16h a0 = *(const v16h*)(awp + ((size_t)lane << 4));
            const v16h a1 = *(const v16h*)(awp + (((size_t)NCHUNK * 32 + lane) << 4));
#pragma unroll
            for (int j = 0; j < NT; ++j) {
                // B fragment: N = lane%16, K = jj + 16*(lane/16) -> 16 consecutive halfs
                const unsigned int* bp =
                    (const unsigned int*)(xbase + rowS + (half << 5) + (j << 4));
                union { v16h h; unsigned int u[8]; } bf;
#pragma unroll
                for (int q = 0; q < 8; ++q) bf.u[q] = bp[q];
                acc[2*j]   = __builtin_amdgcn_wmma_f32_16x16x32_f16(
                                 false, a0, false, bf.h, (short)0, acc[2*j],   false, false);
                acc[2*j+1] = __builtin_amdgcn_wmma_f32_16x16x32_f16(
                                 false, a1, false, bf.h, (short)0, acc[2*j+1], false, false);
            }
            awp += 512;                                  // next chunk (32 lanes * 16 halfs)
        }
        if (++kw == KSZ) { kw = 0; if (++kh == KSZ) { kh = 0; ++ci; } }
    }

    // D layout: N = lane%16 (t'), M = r + 8*(lane/16) (channel within mtile)
    size_t outb = ((size_t)(b * NSPAT + s) * TP) << 5;
#pragma unroll
    for (int j = 0; j < NT; ++j) {
        int tpv = t0 + (j << 4) + (lane & 15);
        if (tpv < TP) {
            float* dst = pot + outb + ((size_t)tpv << 5) + (8 * (lane >> 4));
            *(v8f*)dst        = acc[2*j];      // channels [8*(lane/16), +8)
            *(v8f*)(dst + 16) = acc[2*j+1];    // channels [16 + 8*(lane/16), +8)
        }
    }
}

// ---------------- kernel 5: winner-takes-all sequential scan ----------------
__global__ __launch_bounds__(256) void sfl_wta(const float* __restrict__ pot,
                                               float* __restrict__ out) {
    int idx = blockIdx.x * 256 + threadIdx.x;            // one thread per (b, n)
    if (idx >= B_ * NSPAT) return;
    int b = idx / NSPAT, s = idx % NSPAT;
    int nx = s / HP, ny = s % HP;
    const float* prow = pot + ((size_t)idx) * TP * COUT;
    int dep = 0;
    for (int t = 0; t < TP; ++t) {
        if (dep == 0) {
            const float4* p4 = (const float4*)(prow + ((size_t)t << 5));
            float best = -1.0f; int bc = 0;
#pragma unroll
            for (int q = 0; q < 8; ++q) {
                float4 v = p4[q];
                if (v.x > best) { best = v.x; bc = 4 * q + 0; }
                if (v.y > best) { best = v.y; bc = 4 * q + 1; }
                if (v.z > best) { best = v.z; bc = 4 * q + 2; }
                if (v.w > best) { best = v.w; bc = 4 * q + 3; }
            }
            if (best > THETA) {
                out[((((size_t)b * COUT + bc) * HP + nx) * HP + ny) * TP + t] = 1.0f;
                dep = FODEP - 1;                         // 47
            }
        } else {
            dep--;
        }
    }
}

extern "C" void kernel_launch(void* const* d_in, const int* in_sizes, int n_in,
                              void* d_out, int out_size, void* d_ws, size_t ws_size,
                              hipStream_t stream) {
    const float* x      = (const float*)d_in[0];   // input_spikes [16,2,64,64,100] f32
    const float* weight = (const float*)d_in[1];   // weight [32,2,3,3] f32
    float* out = (float*)d_out;                    // [16,32,31,31,149] f32
    char*  ws  = (char*)d_ws;
    if (ws_size < WS_NEED) return;

    _Float16* aw  = (_Float16*)(ws + AW_OFF);
    _Float16* xh  = (_Float16*)(ws + XH_OFF);
    float*    pot = (float*)(ws + POT_OFF);

    // 1) weight -> fp16 A fragments (36,864 halfs)
    sfl_build_wk<<<144, 256, 0, stream>>>(weight, aw);

    // 2) input -> fp16, zero-padded in time, two parity copies
    long long pad_total = 2ll * (long long)CSTRIDE;      // 58,720,256 halfs
    sfl_pad_input<<<(unsigned)((pad_total + 255) / 256), 256, 0, stream>>>(x, xh, pad_total);

    // 3) zero d_out (winners are one-hot sparse)
    int n4 = out_size / 4;
    sfl_zero<<<(n4 + 255) / 256, 256, 0, stream>>>((float4*)out, n4);

    // 4) WMMA implicit-GEMM conv: 16*961*2 = 30,752 waves, 8 waves/block
    sfl_conv_wmma<<<(B_ * NSPAT * TGROUPS) / 8, 256, 0, stream>>>(xh, aw, pot);

    // 5) winner-takes-all scan: 15,376 threads
    sfl_wta<<<(B_ * NSPAT + 255) / 256, 256, 0, stream>>>(pot, out);
}
